// CrossGroupedQueryAttention_66468913873568
// MI455X (gfx1250) — compile-verified, matrix operand order checked by reading, not run on hardware
//
#include <hip/hip_runtime.h>
#include <math.h>
#include <stdint.h>

// ---------------------------------------------------------------------------
// Types for CDNA5 WMMA (wave32): D(16x16 f32) = A(16x32 bf16) x B(32x16 bf16) + C
// ---------------------------------------------------------------------------
typedef __attribute__((ext_vector_type(16))) __bf16      v16bf;
typedef __attribute__((ext_vector_type(8)))  float       v8f;
typedef __attribute__((ext_vector_type(4)))  unsigned int v4u;

union B16F { v4u q[2]; v16bf v; };

static __device__ __forceinline__ unsigned short f2bf(float f) {
  unsigned int u = __float_as_uint(f);
  u += 0x7fffu + ((u >> 16) & 1u);          // round-to-nearest-even
  return (unsigned short)(u >> 16);
}
static __device__ __forceinline__ float bf2f(unsigned short h) {
  return __uint_as_float(((unsigned int)h) << 16);
}

// Async global->LDS copy (CDNA5, tracked by ASYNCcnt). lds = LDS byte offset,
// g = 64-bit global address. 16 bytes per active lane.
static __device__ __forceinline__ void async_ld_b128(unsigned lds, const void* g) {
  asm volatile("global_load_async_to_lds_b128 %0, %1, off"
               :: "v"(lds), "v"((unsigned long long)(uintptr_t)g)
               : "memory");
}
static __device__ __forceinline__ void wait_async0() {
  asm volatile("s_wait_asynccnt 0x0" ::: "memory");
}

#define DIM   384
#define KVD   192
#define HW    16384          // 128*128
#define BATCH 8
#define PTOT  ((size_t)BATCH * HW)   // 131072 pixels

// ---------------------------------------------------------------------------
// fp32 -> bf16 weight conversion
// ---------------------------------------------------------------------------
__global__ void cvt_bf16_kernel(const float* __restrict__ src,
                                unsigned short* __restrict__ dst, int n) {
  int i = blockIdx.x * 256 + threadIdx.x;
  if (i < n) dst[i] = f2bf(src[i]);
}

__global__ void zero_kernel(unsigned int* __restrict__ p, unsigned int n) {
  unsigned int i = blockIdx.x * 256 + threadIdx.x;
  if (i < n) p[i] = 0u;
}

// ---------------------------------------------------------------------------
// Channel LayerNorm over 384 channels of NCHW, emit bf16 activation [pixel][ch]
// One thread per pixel; channel loads are pixel-coalesced (stride HW).
// ---------------------------------------------------------------------------
__global__ __launch_bounds__(256) void ln_bf16_kernel(
    const float* __restrict__ x, const float* __restrict__ w,
    const float* __restrict__ bsh, unsigned short* __restrict__ out) {
  __shared__ float sw[DIM], sb[DIM];
  for (int i = threadIdx.x; i < DIM; i += 256) { sw[i] = w[i]; sb[i] = bsh[i]; }
  __syncthreads();
  size_t g = (size_t)blockIdx.x * 256 + threadIdx.x;     // pixel id
  size_t b = g >> 14;
  size_t p = g & (HW - 1);
  const float* base = x + (b * DIM) * (size_t)HW + p;
  float s = 0.f, ss = 0.f;
  for (int c = 0; c < DIM; ++c) {
    float v = base[(size_t)c << 14];
    s += v; ss += v * v;
  }
  float mu   = s * (1.f / DIM);
  float var  = ss * (1.f / DIM) - mu * mu;
  float rstd = rsqrtf(var + 1e-5f);
  unsigned short* o = out + g * DIM;
  for (int c = 0; c < DIM; ++c) {
    float v = base[(size_t)c << 14];
    o[c] = f2bf((v - mu) * rstd * sw[c] + sb[c]);
  }
}

// ---------------------------------------------------------------------------
// bf16 WMMA GEMM: out[b][oc][p] (fp32) = act[g][k] * wt[oc][k],  K = 384
// Block tile 128(M) x 64(N), 8 waves, each wave a 16x64 strip (4 accumulators).
// Software-pipelined with async global->LDS copies (ping-pong buffers).
// ---------------------------------------------------------------------------
__global__ __launch_bounds__(256) void gemm_bf16_wmma(
    const unsigned short* __restrict__ act,   // [PTOT][384]
    const unsigned short* __restrict__ wt,    // [OC][384]
    float* __restrict__ out, int OC) {
  __shared__ __align__(16) unsigned short As[2][128 * 32];
  __shared__ __align__(16) unsigned short Bs[2][64 * 32];
  __shared__ __align__(16) float          Cs[128 * 64];

  const int tid  = threadIdx.x;
  const int wave = tid >> 5;
  const int lane = tid & 31;
  const int lr   = lane & 15;
  const int half = (lane >> 4) * 8;            // K offset 0 or 8 per layout
  const size_t g0 = (size_t)blockIdx.x * 128;
  const int n0 = blockIdx.y * 64;
  const int mrow0 = wave * 16;

  const unsigned aOff[2] = { (unsigned)(uintptr_t)&As[0][0],
                             (unsigned)(uintptr_t)&As[1][0] };
  const unsigned bOff[2] = { (unsigned)(uintptr_t)&Bs[0][0],
                             (unsigned)(uintptr_t)&Bs[1][0] };

  // A tile: 128 rows x 32 k = 512 x 16B chunks (2 per thread);
  // B tile:  64 rows x 32 k = 256 x 16B chunks (1 per thread).
  const int arow0 = tid >> 2,           aq0 = (tid & 3) * 8;
  const int arow1 = (tid + 256) >> 2,   aq1 = aq0;            // same qq pattern
  const int brow  = tid >> 2,           bq  = (tid & 3) * 8;

  auto issue_tile = [&](int k0, int buf) {
    async_ld_b128(aOff[buf] + (unsigned)(arow0 * 64 + aq0 * 2),
                  &act[(g0 + arow0) * DIM + k0 + aq0]);
    async_ld_b128(aOff[buf] + (unsigned)(arow1 * 64 + aq1 * 2),
                  &act[(g0 + arow1) * DIM + k0 + aq1]);
    async_ld_b128(bOff[buf] + (unsigned)(brow * 64 + bq * 2),
                  &wt[(size_t)(n0 + brow) * DIM + k0 + bq]);
  };

  issue_tile(0, 0);

  v8f acc[4] = {};
  for (int kt = 0; kt < DIM / 32; ++kt) {
    wait_async0();
    __syncthreads();
    if (kt + 1 < DIM / 32) issue_tile((kt + 1) * 32, (kt + 1) & 1);

    const unsigned short* Ab = As[kt & 1];
    const unsigned short* Bb = Bs[kt & 1];
    B16F a;
    a.q[0] = *(const v4u*)&Ab[(mrow0 + lr) * 32 + half];
    a.q[1] = *(const v4u*)&Ab[(mrow0 + lr) * 32 + 16 + half];
    for (int nt = 0; nt < 4; ++nt) {
      B16F bm;
      bm.q[0] = *(const v4u*)&Bb[(nt * 16 + lr) * 32 + half];
      bm.q[1] = *(const v4u*)&Bb[(nt * 16 + lr) * 32 + 16 + half];
      acc[nt] = __builtin_amdgcn_wmma_f32_16x16x32_bf16(
          false, a.v, false, bm.v, (short)0, acc[nt], false, false);
    }
  }

  __syncthreads();
  for (int nt = 0; nt < 4; ++nt)
    for (int r = 0; r < 8; ++r) {
      int row = mrow0 + r + ((lane >> 4) << 3);
      int col = nt * 16 + lr;
      Cs[row * 64 + col] = acc[nt][r];
    }
  __syncthreads();
  // coalesced store: pixels contiguous per output channel
  for (int v = tid; v < 128 * 64; v += 256) {
    int col = v >> 7, row = v & 127;
    size_t g = g0 + row;
    size_t b = g >> 14, p = g & (HW - 1);
    out[(b * (size_t)OC + n0 + col) * HW + p] = Cs[row * 64 + col];
  }
}

// ---------------------------------------------------------------------------
// Depthwise 3x3 (pad 1) on fp32, emit bf16.
// ---------------------------------------------------------------------------
__global__ void dwconv_q_kernel(const float* __restrict__ in,
                                const float* __restrict__ wdw,
                                unsigned short* __restrict__ qp) {
  size_t idx = (size_t)blockIdx.x * 256 + threadIdx.x;   // b*384*HW
  int p = (int)(idx & (HW - 1));
  size_t bc = idx >> 14;
  int ch = (int)(bc % DIM);
  int y = p >> 7, x = p & 127;
  const float* src = in + bc * (size_t)HW;
  const float* wp  = wdw + ch * 9;
  float acc = 0.f;
  for (int dy = -1; dy <= 1; ++dy) {
    int yy = y + dy; if (yy < 0 || yy > 127) continue;
    for (int dx = -1; dx <= 1; ++dx) {
      int xx = x + dx; if (xx < 0 || xx > 127) continue;
      acc += src[yy * 128 + xx] * wp[(dy + 1) * 3 + (dx + 1)];
    }
  }
  qp[idx] = f2bf(acc);
}

// kv path: ch<96 -> k rows [b][96][p]; ch>=96 -> v transposed [g][128] padded
__global__ void dwconv_kv_kernel(const float* __restrict__ in,
                                 const float* __restrict__ wdw,
                                 unsigned short* __restrict__ kkb,
                                 unsigned short* __restrict__ vvT) {
  size_t idx = (size_t)blockIdx.x * 256 + threadIdx.x;   // b*192*HW
  int p = (int)(idx & (HW - 1));
  size_t bc = idx >> 14;
  int ch = (int)(bc % KVD);
  int b  = (int)(bc / KVD);
  int y = p >> 7, x = p & 127;
  const float* src = in + bc * (size_t)HW;
  const float* wp  = wdw + ch * 9;
  float acc = 0.f;
  for (int dy = -1; dy <= 1; ++dy) {
    int yy = y + dy; if (yy < 0 || yy > 127) continue;
    for (int dx = -1; dx <= 1; ++dx) {
      int xx = x + dx; if (xx < 0 || xx > 127) continue;
      acc += src[yy * 128 + xx] * wp[(dy + 1) * 3 + (dx + 1)];
    }
  }
  unsigned short h = f2bf(acc);
  if (ch < 96) {
    kkb[(((size_t)b * 96 + ch) << 14) + p] = h;
  } else {
    int d = ch - 96;
    int kvh = d / 48, dd = d % 48;
    vvT[((((size_t)b << 14) + p) * 128) + kvh * 64 + dd] = h;
  }
}

// ---------------------------------------------------------------------------
// Row L2-norm reciprocal over HW: scale[r] = 1/max(||row||, 1e-12)
// ---------------------------------------------------------------------------
__global__ __launch_bounds__(256) void rownorm_kernel(
    const unsigned short* __restrict__ rows, float* __restrict__ scale) {
  __shared__ float red[256];
  size_t r = blockIdx.x;
  const unsigned short* src = rows + (r << 14);
  float s = 0.f;
  for (int i = threadIdx.x; i < HW; i += 256) {
    float v = bf2f(src[i]); s += v * v;
  }
  red[threadIdx.x] = s; __syncthreads();
  for (int o = 128; o > 0; o >>= 1) {
    if (threadIdx.x < o) red[threadIdx.x] += red[threadIdx.x + o];
    __syncthreads();
  }
  if (threadIdx.x == 0) scale[r] = 1.f / fmaxf(sqrtf(red[0]), 1e-12f);
}

// ---------------------------------------------------------------------------
// attn logits: [bh][48][48] += q(48 x Kslab) * k^T, K split over blockIdx.y
// ---------------------------------------------------------------------------
__global__ __launch_bounds__(256) void attn_qk_wmma(
    const unsigned short* __restrict__ qp,    // [b][384][HW]
    const unsigned short* __restrict__ kkb,   // [b][96][HW]
    float* __restrict__ attnbuf) {            // [64][48][48]
  int bh = blockIdx.x;
  int b = bh >> 3, head = bh & 7, kvh = head >> 2;
  int wave = threadIdx.x >> 5, lane = threadIdx.x & 31;
  int lr = lane & 15, half = (lane >> 4) * 8;
  int kbase = blockIdx.y * 2048 + wave * 256;

  const unsigned short* qb = qp  + (((size_t)b * DIM + head * 48) << 14);
  const unsigned short* kb = kkb + (((size_t)b * 96  + kvh  * 48) << 14);

  v8f acc[9] = {};
  for (int kc = 0; kc < 8; ++kc) {
    int k0 = kbase + kc * 32;
    B16F a[3], bm[3];
    for (int mt = 0; mt < 3; ++mt) {
      const unsigned short* rp = qb + (((size_t)(mt * 16 + lr)) << 14) + k0;
      a[mt].q[0] = *(const v4u*)(rp + half);
      a[mt].q[1] = *(const v4u*)(rp + 16 + half);
    }
    for (int nt = 0; nt < 3; ++nt) {
      const unsigned short* rp = kb + (((size_t)(nt * 16 + lr)) << 14) + k0;
      bm[nt].q[0] = *(const v4u*)(rp + half);
      bm[nt].q[1] = *(const v4u*)(rp + 16 + half);
    }
    for (int mt = 0; mt < 3; ++mt)
      for (int nt = 0; nt < 3; ++nt)
        acc[mt * 3 + nt] = __builtin_amdgcn_wmma_f32_16x16x32_bf16(
            false, a[mt].v, false, bm[nt].v, (short)0, acc[mt * 3 + nt],
            false, false);
  }
  float* ab = attnbuf + (size_t)bh * 48 * 48;
  for (int mt = 0; mt < 3; ++mt)
    for (int nt = 0; nt < 3; ++nt)
      for (int r = 0; r < 8; ++r) {
        int row = mt * 16 + r + ((lane >> 4) << 3);
        int col = nt * 16 + lr;
        atomicAdd(&ab[row * 48 + col], acc[mt * 3 + nt][r]);
      }
}

// ---------------------------------------------------------------------------
// Fold l2norm scales + temperature, softmax over 48, emit bf16 [bh][48][64]
// (d padded to 64 with zeros for the WMMA K dimension of attn@v).
// ---------------------------------------------------------------------------
__global__ void softmax_kernel(const float* __restrict__ attnbuf,
                               const float* __restrict__ rnq,
                               const float* __restrict__ rnk,
                               const float* __restrict__ temp,
                               unsigned short* __restrict__ attnP) {
  int bh = blockIdx.x;
  int b = bh >> 3, head = bh & 7, kvh = head >> 2;
  int c = threadIdx.x;
  if (c >= 48) return;
  const float* row = attnbuf + ((size_t)bh * 48 + c) * 48;
  float sq = rnq[b * DIM + head * 48 + c];
  float t  = temp[head];
  float vals[48];
  float mx = -1e30f;
  for (int d = 0; d < 48; ++d) {
    float v = row[d] * sq * rnk[b * 96 + kvh * 48 + d] * t;
    vals[d] = v; mx = fmaxf(mx, v);
  }
  float s = 0.f;
  for (int d = 0; d < 48; ++d) { vals[d] = __expf(vals[d] - mx); s += vals[d]; }
  float inv = 1.f / s;
  unsigned short* o = attnP + ((size_t)bh * 48 + c) * 64;
  for (int d = 0; d < 48; ++d) o[d] = f2bf(vals[d] * inv);
  for (int d = 48; d < 64; ++d) o[d] = 0;
}

// ---------------------------------------------------------------------------
// out^T[p][c] = v^T[p][d] * attn[c][d]  -> act3 bf16 [g][384]  (K = 64 padded)
// ---------------------------------------------------------------------------
__global__ __launch_bounds__(256) void attn_av_wmma(
    const unsigned short* __restrict__ attnP,  // [64][48][64]
    const unsigned short* __restrict__ vvT,    // [PTOT][128]
    unsigned short* __restrict__ act3) {       // [PTOT][384]
  int bh = blockIdx.x;
  int b = bh >> 3, head = bh & 7, kvh = head >> 2;
  int wave = threadIdx.x >> 5, lane = threadIdx.x & 31;
  int lr = lane & 15, half = (lane >> 4) * 8;
  size_t pbase = ((size_t)b << 14) + blockIdx.y * 256 + wave * 32;

  v8f acc[2][3] = {};
  for (int k0 = 0; k0 < 64; k0 += 32) {
    B16F a[2], bm[3];
    for (int ms = 0; ms < 2; ++ms) {
      const unsigned short* rp = vvT + (pbase + ms * 16 + lr) * 128 + kvh * 64 + k0;
      a[ms].q[0] = *(const v4u*)(rp + half);
      a[ms].q[1] = *(const v4u*)(rp + 16 + half);
    }
    for (int nt = 0; nt < 3; ++nt) {
      const unsigned short* rp = attnP + ((size_t)bh * 48 + nt * 16 + lr) * 64 + k0;
      bm[nt].q[0] = *(const v4u*)(rp + half);
      bm[nt].q[1] = *(const v4u*)(rp + 16 + half);
    }
    for (int ms = 0; ms < 2; ++ms)
      for (int nt = 0; nt < 3; ++nt)
        acc[ms][nt] = __builtin_amdgcn_wmma_f32_16x16x32_bf16(
            false, a[ms].v, false, bm[nt].v, (short)0, acc[ms][nt],
            false, false);
  }
  for (int ms = 0; ms < 2; ++ms)
    for (int nt = 0; nt < 3; ++nt)
      for (int r = 0; r < 8; ++r) {
        size_t p = pbase + ms * 16 + r + ((lane >> 4) << 3);
        int col = head * 48 + nt * 16 + lr;
        act3[p * DIM + col] = f2bf(acc[ms][nt][r]);
      }
}

// ---------------------------------------------------------------------------
// Host-side orchestration
// ---------------------------------------------------------------------------
extern "C" void kernel_launch(void* const* d_in, const int* in_sizes, int n_in,
                              void* d_out, int out_size, void* d_ws, size_t ws_size,
                              hipStream_t stream) {
  const float* q        = (const float*)d_in[0];
  const float* k        = (const float*)d_in[1];
  const float* w_q      = (const float*)d_in[2];
  const float* w_kv     = (const float*)d_in[3];
  const float* w_q_dw   = (const float*)d_in[4];
  const float* w_kv_dw  = (const float*)d_in[5];
  const float* w_out    = (const float*)d_in[6];
  const float* temp     = (const float*)d_in[7];
  const float* lnq_w    = (const float*)d_in[8];
  const float* lnq_b    = (const float*)d_in[9];
  const float* lnk_w    = (const float*)d_in[10];
  const float* lnk_b    = (const float*)d_in[11];
  float* out            = (float*)d_out;

  char* wsp = (char*)d_ws;
  auto alloc = [&](size_t sz) -> char* {
    char* r = wsp;
    wsp += (sz + 255) & ~(size_t)255;
    return r;
  };
  unsigned short* actQ  = (unsigned short*)alloc(PTOT * DIM * 2);
  unsigned short* actK  = (unsigned short*)alloc(PTOT * DIM * 2);
  unsigned short* wqb   = (unsigned short*)alloc((size_t)DIM * DIM * 2);
  unsigned short* wkvb  = (unsigned short*)alloc((size_t)KVD * DIM * 2);
  unsigned short* woutb = (unsigned short*)alloc((size_t)DIM * DIM * 2);
  float*          C1    = (float*)alloc(PTOT * DIM * 4);
  float*          C2    = (float*)alloc(PTOT * KVD * 4);
  unsigned short* qp    = (unsigned short*)alloc(PTOT * DIM * 2);
  unsigned short* kkb   = (unsigned short*)alloc(PTOT * 96 * 2);
  unsigned short* vvT   = (unsigned short*)alloc(PTOT * 128 * 2);
  float*          rnq   = (float*)alloc((size_t)BATCH * DIM * 4);
  float*          rnk   = (float*)alloc((size_t)BATCH * 96 * 4);
  float*          attnb = (float*)alloc((size_t)64 * 48 * 48 * 4);
  unsigned short* attnP = (unsigned short*)alloc((size_t)64 * 48 * 64 * 2);
  unsigned short* act3  = (unsigned short*)alloc(PTOT * DIM * 2);

  // 1) weights -> bf16
  cvt_bf16_kernel<<<(DIM * DIM + 255) / 256, 256, 0, stream>>>(w_q, wqb, DIM * DIM);
  cvt_bf16_kernel<<<(KVD * DIM + 255) / 256, 256, 0, stream>>>(w_kv, wkvb, KVD * DIM);
  cvt_bf16_kernel<<<(DIM * DIM + 255) / 256, 256, 0, stream>>>(w_out, woutb, DIM * DIM);

  // 2) LayerNorm -> bf16 activations [pixel][ch]
  ln_bf16_kernel<<<PTOT / 256, 256, 0, stream>>>(q, lnq_w, lnq_b, actQ);
  ln_bf16_kernel<<<PTOT / 256, 256, 0, stream>>>(k, lnk_w, lnk_b, actK);

  // 3) 1x1 convs via async-pipelined bf16 WMMA GEMM
  gemm_bf16_wmma<<<dim3(PTOT / 128, DIM / 64), 256, 0, stream>>>(actQ, wqb, C1, DIM);
  gemm_bf16_wmma<<<dim3(PTOT / 128, KVD / 64), 256, 0, stream>>>(actK, wkvb, C2, KVD);

  // 4) zero-fill v^T padding and attn accumulator
  {
    unsigned int nv = (unsigned int)(PTOT * 128 / 2);   // ushorts/2 = uints
    zero_kernel<<<(nv + 255) / 256, 256, 0, stream>>>((unsigned int*)vvT, nv);
    unsigned int na = 64 * 48 * 48;
    zero_kernel<<<(na + 255) / 256, 256, 0, stream>>>((unsigned int*)attnb, na);
  }

  // 5) depthwise 3x3 convs (emit bf16 in attention-friendly layouts)
  dwconv_q_kernel<<<(unsigned int)(PTOT * DIM / 256), 256, 0, stream>>>(C1, w_q_dw, qp);
  dwconv_kv_kernel<<<(unsigned int)(PTOT * KVD / 256), 256, 0, stream>>>(C2, w_kv_dw, kkb, vvT);

  // 6) l2-norm reciprocal scales per (b, channel) row
  rownorm_kernel<<<BATCH * DIM, 256, 0, stream>>>(qp, rnq);
  rownorm_kernel<<<BATCH * 96, 256, 0, stream>>>(kkb, rnk);

  // 7) attn logits (K=16384 split over 8 slabs, f32 atomic combine)
  attn_qk_wmma<<<dim3(64, 8), 256, 0, stream>>>(qp, kkb, attnb);

  // 8) fold norms + temperature, softmax, bf16 attn (d padded to 64)
  softmax_kernel<<<64, 64, 0, stream>>>(attnb, rnq, rnk, temp, attnP);

  // 9) attn @ v  (as out^T = v^T attn^T) -> act3 [pixel][384] bf16
  attn_av_wmma<<<dim3(64, HW / 256), 256, 0, stream>>>(attnP, vvT, act3);

  // 10) final 1x1 conv -> d_out fp32 NCHW
  gemm_bf16_wmma<<<dim3(PTOT / 128, DIM / 64), 256, 0, stream>>>(act3, woutb, out, DIM);
}
